// LiquidTimeConstantBlock_46694884442397
// MI455X (gfx1250) — matrix operand order repaired and checked by reading, not compiled, not run on hardware
//
#include <hip/hip_runtime.h>
#include <hip/hip_bf16.h>

// LiquidTimeConstantBlock for MI455X (gfx1250, wave32, WMMA). B=64, T=2048, D=512.
//
// Phase 0: weights fp32 -> bf16, B-operand-friendly row-major W[j][k].
// Phase 1: pre_mem = x@mem_w.T + mem_b ; pre_tx = x@Wx.T + tau_b as one
//   [131072 x 512] x [512 x 1024] bf16 WMMA GEMM. Block tile 64x1024 so Wc L2
//   traffic is ~2.1GB and x is read from HBM exactly once (~HBM roofline, 23.3TB/s).
// Phase 2: 4 persistent WGs (scan independent per batch row). tau state in LDS
//   (A operand), v state in C-fragment registers, first 128 cols of Wr staged in
//   LDS, rest streamed from the 192MB L2. pre_* loads hoisted above the GEMM;
//   next-step rows prefetched (global_prefetch_b8).

static constexpr int Bdim = 64;
static constexpr int Tdim = 2048;
static constexpr int Ddim = 512;
static constexpr int LDS_LD = 520;          // padded row stride (bf16) -> conflict-free b128 reads

typedef __bf16 bf16_t;
typedef __attribute__((ext_vector_type(16))) __bf16 v16bf;
typedef __attribute__((ext_vector_type(8)))  __bf16 v8bf;
typedef __attribute__((ext_vector_type(8)))  float  v8f;

static __device__ __forceinline__ v16bf cat8(v8bf lo, v8bf hi) {
  return __builtin_shufflevector(lo, hi, 0,1,2,3,4,5,6,7,8,9,10,11,12,13,14,15);
}

// A fragment, 16x32 bf16 (ISA 7.12.2): lane m = lane&15.
// lanes 0-15 hold K {0..7, 16..23}; lanes 16-31 hold K {8..15, 24..31}.
static __device__ __forceinline__ v16bf load_fragA(const bf16_t* base, int ld, int lane, int kc) {
  const int m = lane & 15, ksel = (lane >> 4) & 1;
  const bf16_t* p = base + m * ld + kc * 32 + ksel * 8;
  v8bf lo = *(const v8bf*)p;
  v8bf hi = *(const v8bf*)(p + 16);
  return cat8(lo, hi);
}

// B fragment, 32x16 bf16: lane n = lane&15 is the output column (= row j of W[j][k]);
// lanes 0-15 hold K {0..7, 16..23} of the chunk, lanes 16-31 hold K {8..15, 24..31}.
static __device__ __forceinline__ v16bf load_fragB(const bf16_t* wjbase, int ld, int lane, int kc) {
  const int n = lane & 15, khalf = (lane >> 4) & 1;
  const bf16_t* p = wjbase + n * ld + kc * 32 + khalf * 8;
  v8bf lo = *(const v8bf*)p;
  v8bf hi = *(const v8bf*)(p + 16);
  return cat8(lo, hi);
}

// ---------------- Phase 0: weight conversion ----------------
__global__ __launch_bounds__(256) void ltc_convert_weights(
    const float* __restrict__ tau_w, const float* __restrict__ mem_w,
    bf16_t* __restrict__ Wc, bf16_t* __restrict__ Wr)
{
  const int gid = blockIdx.x * 256 + threadIdx.x;   // 786432 threads exactly
  if (gid < 1024 * 512) {
    const int j = gid >> 9, k = gid & 511;
    const float w = (j < 512) ? mem_w[j * 512 + k] : tau_w[(j - 512) * 1024 + k];
    Wc[gid] = (bf16_t)w;
  } else {
    const int g = gid - 1024 * 512;
    const int j = g >> 9, k = g & 511;
    Wr[g] = (bf16_t)tau_w[j * 1024 + 512 + k];
  }
}

// ---------------- Phase 1: parallel pre-activation GEMM ----------------
// grid = B*T/64 blocks, 512 threads (16 waves). Block tile: 64 rows x 1024 cols.
// Wave w: M-tiles 0..3 x N-tiles {4w..4w+3}. K = 512 (16 WMMA k-chunks).
__global__ __launch_bounds__(512) void ltc_phase1(
    const float* __restrict__ x, const bf16_t* __restrict__ Wc,
    const float* __restrict__ mem_b, const float* __restrict__ tau_b,
    float* __restrict__ pre_mem, float* __restrict__ pre_tx)
{
  __shared__ bf16_t xs[64 * LDS_LD];          // 66.6 KB
  const int tid = threadIdx.x;
  const long r0 = (long)blockIdx.x * 64;

  // stage x tile fp32 -> bf16 into LDS, coalesced (col = tid)
  #pragma unroll 8
  for (int i = 0; i < 64; ++i)
    xs[i * LDS_LD + tid] = (bf16_t)x[(r0 + i) * Ddim + tid];
  __syncthreads();

  const int wave = tid >> 5, lane = tid & 31;

  const v8f vzero = {0.f,0.f,0.f,0.f,0.f,0.f,0.f,0.f};
  v8f c[4][4];                                 // [mt][q]
  #pragma unroll
  for (int mt = 0; mt < 4; ++mt)
    #pragma unroll
    for (int q = 0; q < 4; ++q) c[mt][q] = vzero;

  #pragma unroll 2
  for (int kc = 0; kc < 16; ++kc) {
    v16bf a[4], b[4];
    #pragma unroll
    for (int mt = 0; mt < 4; ++mt)
      a[mt] = load_fragA(xs + mt * 16 * LDS_LD, LDS_LD, lane, kc);
    #pragma unroll
    for (int q = 0; q < 4; ++q)
      b[q] = load_fragB(Wc + (long)((wave * 4 + q) * 16) * Ddim, Ddim, lane, kc);
    #pragma unroll
    for (int mt = 0; mt < 4; ++mt)
      #pragma unroll
      for (int q = 0; q < 4; ++q)
        c[mt][q] = __builtin_amdgcn_wmma_f32_16x16x32_bf16(
            false, a[mt], false, b[q], (short)0, c[mt][q], false, false);
  }

  // epilogue: add bias, scatter fp32 (full fidelity for the scan)
  const int n = lane & 15;
  const int mBase = (lane >> 4) * 8;           // C layout: lanes>=16 hold rows M=8..15
  #pragma unroll
  for (int q = 0; q < 4; ++q) {
    const int jcol = (wave * 4 + q) * 16 + n;  // 0..1023, tile-uniform split at 512
    float bias; float* dst; int col;
    if (jcol < 512) { bias = mem_b[jcol];       dst = pre_mem; col = jcol; }
    else            { bias = tau_b[jcol - 512]; dst = pre_tx;  col = jcol - 512; }
    #pragma unroll
    for (int mt = 0; mt < 4; ++mt)
      #pragma unroll
      for (int i = 0; i < 8; ++i)
        dst[(r0 + mt * 16 + mBase + i) * Ddim + col] = c[mt][q][i] + bias;
  }
}

// ---------------- Phase 2: persistent recurrent scan ----------------
// grid = 4 blocks (B/16), 512 threads (16 waves, 2 N-tiles each => all 512 cols).
static constexpr int WLDS_COLS = 128;          // cols 0..127 of Wr staged in LDS (waves 0..3)

__global__ __launch_bounds__(512) void ltc_phase2(
    const float* __restrict__ pre_mem, const float* __restrict__ pre_tx,
    const bf16_t* __restrict__ Wr, const float* __restrict__ log_thresh,
    float* __restrict__ spikes, float* __restrict__ tau_out, float* __restrict__ v_out)
{
  __shared__ bf16_t taus[16 * LDS_LD];               // 16.6 KB
  __shared__ bf16_t wlds[WLDS_COLS * LDS_LD];        // 133.1 KB  (total ~150KB LDS)
  const int tid = threadIdx.x;

  // init carry tau = 1
  #pragma unroll
  for (int i = 0; i < 16; ++i) taus[i * LDS_LD + tid] = (bf16_t)1.0f;
  // stage Wr[0..127][*] into LDS (16B chunks, padded stride)
  for (int it = tid; it < WLDS_COLS * (Ddim / 8); it += 512) {
    const int j = it >> 6, kk = (it & 63) * 8;
    *(v8bf*)&wlds[j * LDS_LD + kk] = *(const v8bf*)&Wr[j * Ddim + kk];
  }
  __syncthreads();

  const int wave = tid >> 5, lane = tid & 31;
  const int n = lane & 15, mBase = (lane >> 4) * 8;
  const int b0 = blockIdx.x * 16;
  const bool b_in_lds = (wave < (WLDS_COLS / 32));   // waves 0..3 -> cols 0..127

  float thr[2];
  long rowbase[2][8];
  #pragma unroll
  for (int q = 0; q < 2; ++q) {
    const int j = (wave * 2 + q) * 16 + n;
    thr[q] = 1.0f / (1.0f + __expf(-log_thresh[j]));
    #pragma unroll
    for (int i = 0; i < 8; ++i)
      rowbase[q][i] = (long)(b0 + mBase + i) * Tdim * Ddim + j;
  }

  const v8f vzero = {0.f,0.f,0.f,0.f,0.f,0.f,0.f,0.f};
  v8f v[2]    = {vzero, vzero};     // membrane state in registers
  v8f tauv[2] = {vzero, vzero};

  for (int t = 0; t < Tdim; ++t) {
    // hoist activation loads: latency hidden under the GEMM below
    float pm[2][8], px[2][8];
    #pragma unroll
    for (int q = 0; q < 2; ++q)
      #pragma unroll
      for (int i = 0; i < 8; ++i) {
        const long base = rowbase[q][i] + (long)t * Ddim;
        px[q][i] = pre_tx[base];
        pm[q][i] = pre_mem[base];
      }
    if (t + 1 < Tdim) {   // prefetch next step's rows (global_prefetch_b8)
      #pragma unroll
      for (int q = 0; q < 2; ++q) {
        __builtin_prefetch(&pre_tx[rowbase[q][0] + (long)(t + 1) * Ddim], 0, 1);
        __builtin_prefetch(&pre_mem[rowbase[q][0] + (long)(t + 1) * Ddim], 0, 1);
      }
    }

    // A = tau_t (shared across all 16 waves)
    v16bf a[16];
    #pragma unroll
    for (int kc = 0; kc < 16; ++kc) a[kc] = load_fragA(taus, LDS_LD, lane, kc);
    __syncthreads();   // all reads done before anyone overwrites taus

    v8f c[2] = {vzero, vzero};
    #pragma unroll
    for (int kc = 0; kc < 16; ++kc) {
      #pragma unroll
      for (int q = 0; q < 2; ++q) {
        const int jt16 = (wave * 2 + q) * 16;
        v16bf bfr;
        if (b_in_lds) bfr = load_fragB(wlds + jt16 * LDS_LD, LDS_LD, lane, kc);
        else          bfr = load_fragB(Wr + (long)jt16 * Ddim, Ddim, lane, kc);
        c[q] = __builtin_amdgcn_wmma_f32_16x16x32_bf16(
            false, a[kc], false, bfr, (short)0, c[q], false, false);
      }
    }

    // elementwise scan step in fp32
    #pragma unroll
    for (int q = 0; q < 2; ++q) {
      #pragma unroll
      for (int i = 0; i < 8; ++i) {
        const long base  = rowbase[q][i] + (long)t * Ddim;
        const float z     = c[q][i] + px[q][i];                // + x@Wx.T + tau_b
        const float tau   = 1.0f / (1.0f + __expf(-z));
        const float alpha = __expf(-1.0f / (tau + 1e-6f));
        const float vn    = alpha * v[q][i] + (1.0f - alpha) * pm[q][i];
        const float s     = (vn >= thr[q]) ? 1.0f : 0.0f;
        spikes[base] = s;
        v[q][i]      = vn * (1.0f - s);
        tauv[q][i]   = tau;
      }
    }

    // publish tau_{t+1}
    #pragma unroll
    for (int q = 0; q < 2; ++q) {
      const int j = (wave * 2 + q) * 16 + n;
      #pragma unroll
      for (int i = 0; i < 8; ++i)
        taus[(mBase + i) * LDS_LD + j] = (bf16_t)tauv[q][i];
    }
    __syncthreads();
  }

  // final carry outputs
  #pragma unroll
  for (int q = 0; q < 2; ++q) {
    const int j = (wave * 2 + q) * 16 + n;
    #pragma unroll
    for (int i = 0; i < 8; ++i) {
      const int b = b0 + mBase + i;
      tau_out[b * Ddim + j] = tauv[q][i];
      v_out[b * Ddim + j]   = v[q][i];
    }
  }
}

// ---------------- host launcher ----------------
extern "C" void kernel_launch(void* const* d_in, const int* in_sizes, int n_in,
                              void* d_out, int out_size, void* d_ws, size_t ws_size,
                              hipStream_t stream) {
  const float* x          = (const float*)d_in[0];
  const float* tau_w      = (const float*)d_in[1];
  const float* tau_b      = (const float*)d_in[2];
  const float* mem_w      = (const float*)d_in[3];
  const float* mem_b      = (const float*)d_in[4];
  const float* log_thresh = (const float*)d_in[5];

  const size_t BTD = (size_t)Bdim * Tdim * Ddim;
  float* spikes  = (float*)d_out;
  float* tau_out = spikes + BTD;
  float* v_out   = tau_out + (size_t)Bdim * Ddim;

  float*  pre_mem = (float*)d_ws;
  float*  pre_tx  = pre_mem + BTD;
  bf16_t* Wc      = (bf16_t*)(pre_tx + BTD);
  bf16_t* Wr      = Wc + (size_t)1024 * 512;

  ltc_convert_weights<<<3072, 256, 0, stream>>>(tau_w, mem_w, Wc, Wr);
  ltc_phase1<<<(Bdim * Tdim) / 64, 512, 0, stream>>>(x, Wc, mem_b, tau_b, pre_mem, pre_tx);
  ltc_phase2<<<Bdim / 16, 512, 0, stream>>>(pre_mem, pre_tx, Wr, log_thresh,
                                            spikes, tau_out, v_out);
}
